// MultiscaleDeformableAttention_22582938042588
// MI455X (gfx1250) — matrix operand order repaired and checked by reading, not compiled, not run on hardware
//
#include <hip/hip_runtime.h>
#include <hip/hip_bf16.h>
#include <math.h>

// ---------------------------------------------------------------------------
// MI455X (gfx1250) Multiscale Deformable Attention
//   - 4x GEMM via v_wmma_f32_16x16x32_f16, K=256 fully unrolled,
//     weights pre-packed to f16 B-fragments (2x global_load_b128 per frag),
//     2 N-tiles per wave reusing one A fragment.
//   - sampling: float4 gathers, wave-shfl distributed offsets/softmax.
// ---------------------------------------------------------------------------

typedef __attribute__((ext_vector_type(16))) _Float16 v16h;
typedef __attribute__((ext_vector_type(8)))  float    v8f;
typedef __attribute__((ext_vector_type(4)))  float    v4f;

#define BQ      43520      // B*Q == B*TOTAL_TOKENS
#define QN      21760
#define EMBED   256

// ---------------------------------------------------------------------------
// Pack W[K=256, N] f32 (row-major) into f16 B-fragments:
//   Wp[((tileN*8 + g)*32 + lane)*16 + e] = W[(g*32 + (lane>>4)*16 + e)*N
//                                            + tileN*16 + (lane&15)]
// so a wave reads one 32x16 fragment as a contiguous 32B/lane chunk.
// ---------------------------------------------------------------------------
__global__ void pack_b_kernel(const float* __restrict__ W,
                              _Float16* __restrict__ Wp, int N)
{
    const int t = blockIdx.x * blockDim.x + threadIdx.x; // (tileN*8+g)*32+lane
    const int total = (N / 16) * 8 * 32;
    if (t >= total) return;
    const int lane  = t & 31;
    const int gt    = t >> 5;
    const int g     = gt & 7;        // K/32 == 8 groups
    const int tileN = gt >> 3;
    const int nCol  = tileN * 16 + (lane & 15);
    const int kb    = g * 32 + (lane >> 4) * 16;
    _Float16* dst = Wp + (size_t)t * 16;
#pragma unroll
    for (int e = 0; e < 16; ++e)
        dst[e] = (_Float16)W[(size_t)(kb + e) * N + nCol];
}

// ---------------------------------------------------------------------------
// C[M,N] = A[M,256] @ W[256,N] + bias[N]; W pre-packed (Wp).
// 256 threads = 8 waves; wave w owns N-tiles w*TILES .. w*TILES+TILES-1.
// TILES*8*16 must equal N (N=256 -> TILES=2, N=128 -> TILES=1): no divergence.
// ---------------------------------------------------------------------------
template <int TILES>
__global__ __launch_bounds__(256) void gemm_wmma_f16(
    const float* __restrict__ A, const _Float16* __restrict__ Wp,
    const float* __restrict__ bias, float* __restrict__ C, int M, int N)
{
    const int wave  = threadIdx.x >> 5;
    const int lane  = threadIdx.x & 31;
    const int tileM = blockIdx.x;

    // A fragment (contiguous k order, same as Wp packing): lanes 0-15 K 0-15, lanes 16-31 K 16-31
    const int mRow = tileM * 16 + (lane & 15);
    const int aK   = (lane >> 4) * 16;
    const float* aRow = A + (size_t)mRow * 256;

    const _Float16* bp[TILES];
    v8f acc[TILES];
    const v8f zero = {};
#pragma unroll
    for (int t = 0; t < TILES; ++t) {
        const int tN = wave * TILES + t;
        bp[t]  = Wp + ((size_t)(tN * 8) * 32 + lane) * 16;
        acc[t] = zero;
    }

#pragma unroll
    for (int g = 0; g < 8; ++g) {                 // K = 256, steps of 32
        const int k0 = g * 32;
        v4f a0 = *(const v4f*)(aRow + k0 + aK);
        v4f a1 = *(const v4f*)(aRow + k0 + aK + 4);
        v4f a2 = *(const v4f*)(aRow + k0 + aK + 8);
        v4f a3 = *(const v4f*)(aRow + k0 + aK + 12);
        v16h af;
#pragma unroll
        for (int i = 0; i < 4; ++i) {
            af[i]      = (_Float16)a0[i];
            af[4 + i]  = (_Float16)a1[i];
            af[8 + i]  = (_Float16)a2[i];
            af[12 + i] = (_Float16)a3[i];
        }
#pragma unroll
        for (int t = 0; t < TILES; ++t) {
            v16h bf = *(const v16h*)(bp[t] + (size_t)g * (32 * 16));
            acc[t] = __builtin_amdgcn_wmma_f32_16x16x32_f16(
                         false, af, false, bf, (short)0, acc[t], false, false);
        }
    }

    // C/D layout: VGPR r -> M = mBase + r; N = lane&15 within tile
    const int mBase = tileM * 16 + (lane >> 4) * 8;
#pragma unroll
    for (int t = 0; t < TILES; ++t) {
        const int nCol = (wave * TILES + t) * 16 + (lane & 15);
        const float bv = bias[nCol];
        float* cp = C + (size_t)mBase * N + nCol;
#pragma unroll
        for (int r = 0; r < 8; ++r)
            cp[(size_t)r * N] = acc[t][r] + bv;
    }
}

// Collapse W_out over heads: W_sum[c,d] = sum_h W_out[h,c,d]; same for bias.
__global__ void reduce_wout_kernel(const float* __restrict__ W_out,
                                   const float* __restrict__ b_out,
                                   float* __restrict__ W_sum,
                                   float* __restrict__ b_sum)
{
    int idx = blockIdx.x * blockDim.x + threadIdx.x;
    if (idx < 256 * 256) {
        float s = 0.f;
#pragma unroll
        for (int h = 0; h < 8; ++h) s += W_out[h * 65536 + idx];
        W_sum[idx] = s;
    } else if (idx < 256 * 256 + 256) {
        int j = idx - 65536;
        float s = 0.f;
#pragma unroll
        for (int h = 0; h < 8; ++h) s += b_out[h * 256 + j];
        b_sum[j] = s;
    }
}

// ---------------------------------------------------------------------------
// Sampling + softmax + aggregation. One 64-thread block per (b,q).
// Thread: head h = tid>>3, channel quad d4 = (tid&7)*4 -> float4 gathers.
// Offsets/logits loaded cooperatively (1 b128 + 2 b32 per lane) and shared
// within the 8-lane head group via __shfl; 8 lanes x 16B = coalesced 128B.
// agg aliases P_off: each thread reads exactly the 4 floats it later writes.
// ---------------------------------------------------------------------------
__global__ __launch_bounds__(64) void msda_sample_kernel(
    const float* __restrict__ P_off,   // [BQ, 256] sampling offsets (+bias)
    const float* __restrict__ P_attn,  // [BQ, 128] attention logits (+bias)
    const float* __restrict__ refpts,  // [BQ, 2]
    const float* __restrict__ v,       // [BQ, 256] projected values (b-major)
    float* __restrict__ agg)           // [BQ, 256] (may alias P_off)
{
    const int Hs[4]     = {128, 64, 32, 16};
    const int Wls[4]    = {128, 64, 32, 16};
    const int starts[4] = {0, 16384, 20480, 21504};

    const int row  = blockIdx.x;          // b*Q + q
    const int b    = row / QN;
    const int tid  = threadIdx.x;         // 0..63
    const int lane = tid & 31;
    const int h    = tid >> 3;            // global head 0..7
    const int d4   = (tid & 7) * 4;       // channel quad within head
    const int j    = lane & 7;            // lane within head group
    const int grp  = lane & 24;           // first lane of this head group

    // Cooperative loads: 4 offsets + 2 logits per lane (head-local).
    v4f offq = *(const v4f*)(P_off + (size_t)row * 256 + h * 32 + j * 4);
    const float o0 = offq[0], o1 = offq[1], o2 = offq[2], o3 = offq[3];
    const float l0 = P_attn[(size_t)row * 128 + h * 16 + j * 2 + 0];
    const float l1 = P_attn[(size_t)row * 128 + h * 16 + j * 2 + 1];

    // Softmax over the 16 logits of this head (8-lane group reduction).
    float m = fmaxf(l0, l1);
#pragma unroll
    for (int s = 1; s < 8; s <<= 1) m = fmaxf(m, __shfl_xor(m, s, 32));
    const float e0 = __expf(l0 - m), e1 = __expf(l1 - m);
    float ps = e0 + e1;
#pragma unroll
    for (int s = 1; s < 8; s <<= 1) ps += __shfl_xor(ps, s, 32);
    const float inv = 1.0f / ps;

    const float refx = refpts[(size_t)row * 2 + 0];
    const float refy = refpts[(size_t)row * 2 + 1];

    v4f acc = {};
#pragma unroll
    for (int l = 0; l < 4; ++l) {
        const int Hl = Hs[l], Wl = Wls[l];
        const float* vb = v + (size_t)(b * QN + starts[l]) * EMBED + h * 32 + d4;
#pragma unroll
        for (int k = 0; k < 4; ++k) {
            const int lk  = l * 4 + k;
            const int src = grp + (lk >> 1);
            // offsets lk*2, lk*2+1 live in (o0,o1) for even lk, (o2,o3) for odd
            const float ox = __shfl((lk & 1) ? o2 : o0, src, 32);
            const float oy = __shfl((lk & 1) ? o3 : o1, src, 32);
            const float aw = __shfl((lk & 1) ? e1 : e0, src, 32) * inv;

            float ax = fminf(fmaxf(refx + ox, 0.f), 1.f);
            float ay = fminf(fmaxf(refy + oy, 0.f), 1.f);
            float x  = ax * (float)Wl - 0.5f;
            float y  = ay * (float)Hl - 0.5f;
            float xf = floorf(x), yf = floorf(y);
            float lx = x - xf, ly = y - yf;
            int   x0 = (int)xf, y0 = (int)yf;
#pragma unroll
            for (int c = 0; c < 4; ++c) {
                const int yi = y0 + (c >> 1);
                const int xi = x0 + (c & 1);
                const float cw = ((c & 1) ? lx : 1.f - lx) *
                                 ((c >> 1) ? ly : 1.f - ly);
                const bool valid = (yi >= 0) & (yi < Hl) & (xi >= 0) & (xi < Wl);
                const int yc = yi < 0 ? 0 : (yi > Hl - 1 ? Hl - 1 : yi);
                const int xc = xi < 0 ? 0 : (xi > Wl - 1 ? Wl - 1 : xi);
                v4f g = *(const v4f*)(vb + (size_t)(yc * Wl + xc) * EMBED);
                const float wS = valid ? aw * cw : 0.0f;
                acc[0] += wS * g[0];
                acc[1] += wS * g[1];
                acc[2] += wS * g[2];
                acc[3] += wS * g[3];
            }
        }
    }
    *(v4f*)(agg + (size_t)row * 256 + h * 32 + d4) = acc;
}

extern "C" void kernel_launch(void* const* d_in, const int* in_sizes, int n_in,
                              void* d_out, int out_size, void* d_ws, size_t ws_size,
                              hipStream_t stream) {
    const float* queries = (const float*)d_in[0];
    const float* refpts  = (const float*)d_in[1];
    const float* value   = (const float*)d_in[2];
    const float* W_off   = (const float*)d_in[3];
    const float* b_off   = (const float*)d_in[4];
    const float* W_attn  = (const float*)d_in[5];
    const float* b_attn  = (const float*)d_in[6];
    const float* W_val   = (const float*)d_in[7];
    const float* b_val   = (const float*)d_in[8];
    const float* W_out   = (const float*)d_in[9];
    const float* b_out   = (const float*)d_in[10];
    float* out = (float*)d_out;

    const int M = BQ;                             // 43520 rows = 2720 M-tiles
    float* ws     = (float*)d_ws;
    float* P_off  = ws;                           // M*256 f32
    float* P_attn = P_off + (size_t)M * 256;      // M*128 f32
    float* Vproj  = P_attn + (size_t)M * 128;     // M*256 f32
    float* W_sum  = Vproj + (size_t)M * 256;      // 65536 f32
    float* b_sum  = W_sum + 65536;                // 256 f32
    _Float16* Wp_off  = (_Float16*)(b_sum + 256);     // 256*256 f16
    _Float16* Wp_attn = Wp_off + 256 * 256;           // 256*128 f16
    _Float16* Wp_val  = Wp_attn + 256 * 128;          // 256*256 f16
    _Float16* Wp_sum  = Wp_val + 256 * 256;           // 256*256 f16
    float* agg = P_off;                           // aliased (safe, see kernel)

    dim3 blk(256);

    // Weight preprocessing (all tiny).
    reduce_wout_kernel<<<(65536 + 256 + 255) / 256, blk, 0, stream>>>(
        W_out, b_out, W_sum, b_sum);
    pack_b_kernel<<<(4096 + 255) / 256, blk, 0, stream>>>(W_off, Wp_off, 256);
    pack_b_kernel<<<(2048 + 255) / 256, blk, 0, stream>>>(W_attn, Wp_attn, 128);
    pack_b_kernel<<<(4096 + 255) / 256, blk, 0, stream>>>(W_val, Wp_val, 256);
    pack_b_kernel<<<(4096 + 255) / 256, blk, 0, stream>>>(W_sum, Wp_sum, 256);

    // P_off = queries @ W_off + b_off            [43520 x 256]
    gemm_wmma_f16<2><<<dim3(M / 16), blk, 0, stream>>>(
        queries, Wp_off, b_off, P_off, M, 256);
    // P_attn = queries @ W_attn + b_attn         [43520 x 128]
    gemm_wmma_f16<1><<<dim3(M / 16), blk, 0, stream>>>(
        queries, Wp_attn, b_attn, P_attn, M, 128);
    // Vproj = value @ W_val + b_val              [43520 x 256]
    gemm_wmma_f16<2><<<dim3(M / 16), blk, 0, stream>>>(
        value, Wp_val, b_val, Vproj, M, 256);

    // Softmax + bilinear sampling + aggregation -> agg (overwrites P_off)
    msda_sample_kernel<<<dim3(M), dim3(64), 0, stream>>>(
        P_off, P_attn, refpts, Vproj, agg);

    // out = agg @ (sum_h W_out[h]) + sum_h b_out[h]
    gemm_wmma_f16<2><<<dim3(M / 16), blk, 0, stream>>>(
        agg, Wp_sum, b_sum, out, M, 256);
}
